// LSPM_41455024341635
// MI455X (gfx1250) — compile-verified, matrix operand order checked by reading, not run on hardware
//
#include <hip/hip_runtime.h>
#include <hip/hip_bf16.h>

typedef __attribute__((ext_vector_type(16))) __bf16 bf16x16;
typedef __attribute__((ext_vector_type(8)))  float  floatx8;
typedef int v4i_ __attribute__((vector_size(16)));
typedef __attribute__((address_space(1))) v4i_* gptr_v4i;
typedef __attribute__((address_space(3))) v4i_* lptr_v4i;

#define NB 4
#define CCH 256
#define NT 2304   // 48*48
#define KPAD 320  // 256 + 50 padded to multiple of 32

#define WMMA_BF16(a, b, c) \
  __builtin_amdgcn_wmma_f32_16x16x32_bf16(false, (a), false, (b), (short)0, (c), false, false)

#if defined(__gfx1250__) && __has_builtin(__builtin_amdgcn_global_load_async_to_lds_b128) && \
    __has_builtin(__builtin_amdgcn_s_wait_asynccnt)
#define LSPM_ASYNC_LDS 1
#endif

// ---------------------------------------------------------------------------
// 1. convert: x fp32 [B][C][N] -> xT bf16 [B][N][C] and Z bf16 rows 0..255
// ---------------------------------------------------------------------------
__global__ __launch_bounds__(256) void lspm_convert(const float* __restrict__ x,
                                                    __bf16* __restrict__ xT,
                                                    __bf16* __restrict__ Z) {
  size_t idx = (size_t)blockIdx.x * 256 + threadIdx.x;   // over B*C*N, n fastest
  int n = idx % NT;
  int c = (idx / NT) % CCH;
  int b = idx / ((size_t)NT * CCH);
  float v = x[idx];
  __bf16 bv = (__bf16)v;
  xT[((size_t)b * NT + n) * CCH + c] = bv;               // transposed copy (K=C contig)
  Z[((size_t)b * KPAD + c) * NT + n] = bv;               // row-major copy
}

// ---------------------------------------------------------------------------
// 2. Gram: scores[b] = xT[b] (N x C) * xT[b]^T  -> [B][N][N] fp32
//    block = 128x128 tile, 8 waves, each wave 32x64 (2x4 wmma frags)
//    Double-buffered LDS; async global->LDS DMA (ASYNCcnt) when available.
// ---------------------------------------------------------------------------
__global__ __launch_bounds__(256) void lspm_gram(const __bf16* __restrict__ xT,
                                                 float* __restrict__ scores) {
  const int b  = blockIdx.z;
  const int n0 = blockIdx.y * 128;
  const int m0 = blockIdx.x * 128;
  __shared__ __bf16 As[2][128][32];
  __shared__ __bf16 Bs[2][128][32];
  const int t = threadIdx.x, wave = t >> 5, lane = t & 31;
  const int waveM = wave & 3, waveN = wave >> 2;
  const int half = lane >> 4, l = lane & 15;
  floatx8 acc[2][4];
  for (int i = 0; i < 2; ++i)
    for (int j = 0; j < 4; ++j) acc[i][j] = (floatx8){0,0,0,0,0,0,0,0};
  const __bf16* xTb = xT + (size_t)b * NT * CCH;

#ifdef LSPM_ASYNC_LDS
  // Each thread owns 2 slots of A and 2 slots of B per 128x32 tile (4 async ops).
  auto issue = [&](int buf, int kk) {
    for (int i = 0; i < 2; ++i) {
      int slot = t + i * 256;          // 512 slots = 128 rows x 4 quarters
      int row = slot >> 2, q = (slot & 3) * 8;
      __builtin_amdgcn_global_load_async_to_lds_b128(
          (gptr_v4i)(xTb + (size_t)(n0 + row) * CCH + kk + q),
          (lptr_v4i)(&As[buf][row][q]), 0, 0);
      __builtin_amdgcn_global_load_async_to_lds_b128(
          (gptr_v4i)(xTb + (size_t)(m0 + row) * CCH + kk + q),
          (lptr_v4i)(&Bs[buf][row][q]), 0, 0);
    }
  };
  issue(0, 0);
  for (int kt = 0; kt < 8; ++kt) {
    const int buf = kt & 1;
    if (kt < 7) {
      issue(buf ^ 1, (kt + 1) * 32);
      __builtin_amdgcn_s_wait_asynccnt(4);   // current tile's 4 ops complete (in-order)
    } else {
      __builtin_amdgcn_s_wait_asynccnt(0);
    }
    __syncthreads();                          // all waves' DMA visible in LDS
    bf16x16 af[2], bfr[4];
    for (int mi = 0; mi < 2; ++mi)
      af[mi] = *(const bf16x16*)(&As[buf][waveM * 32 + mi * 16 + l][half * 16]);
    for (int ni = 0; ni < 4; ++ni)
      bfr[ni] = *(const bf16x16*)(&Bs[buf][waveN * 64 + ni * 16 + l][half * 16]);
    for (int mi = 0; mi < 2; ++mi)
      for (int ni = 0; ni < 4; ++ni)
        acc[mi][ni] = WMMA_BF16(af[mi], bfr[ni], acc[mi][ni]);
    __syncthreads();                          // WAR: buffer may be refilled next iter
  }
#else
  for (int kk = 0; kk < CCH; kk += 32) {
    for (int i = 0; i < 2; ++i) {
      int slot = t + i * 256;
      int row = slot >> 2, q = (slot & 3) * 8;
      *(uint4*)(&As[0][row][q]) = *(const uint4*)(xTb + (size_t)(n0 + row) * CCH + kk + q);
      *(uint4*)(&Bs[0][row][q]) = *(const uint4*)(xTb + (size_t)(m0 + row) * CCH + kk + q);
    }
    __syncthreads();
    bf16x16 af[2], bfr[4];
    for (int mi = 0; mi < 2; ++mi)
      af[mi] = *(const bf16x16*)(&As[0][waveM * 32 + mi * 16 + l][half * 16]);
    for (int ni = 0; ni < 4; ++ni)
      bfr[ni] = *(const bf16x16*)(&Bs[0][waveN * 64 + ni * 16 + l][half * 16]);
    for (int mi = 0; mi < 2; ++mi)
      for (int ni = 0; ni < 4; ++ni)
        acc[mi][ni] = WMMA_BF16(af[mi], bfr[ni], acc[mi][ni]);
    __syncthreads();
  }
#endif

  float* out = scores + (size_t)b * NT * NT;
  const int rb = half * 8;
  for (int mi = 0; mi < 2; ++mi)
    for (int ni = 0; ni < 4; ++ni) {
      int col = m0 + waveN * 64 + ni * 16 + l;
      for (int r = 0; r < 8; ++r) {
        int row = n0 + waveM * 32 + mi * 16 + rb + r;
        out[(size_t)row * NT + col] = acc[mi][ni][r];
      }
    }
}

// ---------------------------------------------------------------------------
// 3. softmax per row (length 2304), fp32 in -> bf16 out
// ---------------------------------------------------------------------------
__global__ __launch_bounds__(256) void lspm_softmax(const float* __restrict__ scores,
                                                    __bf16* __restrict__ attn) {
  const size_t row = blockIdx.x;
  const float* srow = scores + row * NT;
  __bf16* arow = attn + row * NT;
  const int t = threadIdx.x;
  __shared__ float red[256];
  float vals[9];
  float m = -3.4e38f;
  for (int i = 0; i < 9; ++i) { vals[i] = srow[t + i * 256]; m = fmaxf(m, vals[i]); }
  red[t] = m; __syncthreads();
  for (int s = 128; s > 0; s >>= 1) { if (t < s) red[t] = fmaxf(red[t], red[t + s]); __syncthreads(); }
  m = red[0]; __syncthreads();
  float sum = 0.f;
  for (int i = 0; i < 9; ++i) { vals[i] = __expf(vals[i] - m); sum += vals[i]; }
  red[t] = sum; __syncthreads();
  for (int s = 128; s > 0; s >>= 1) { if (t < s) red[t] += red[t + s]; __syncthreads(); }
  float inv = 1.f / red[0];
  for (int i = 0; i < 9; ++i) arow[t + i * 256] = (__bf16)(vals[i] * inv);
}

// ---------------------------------------------------------------------------
// 4. xc_all[b][s][n] = sum_c w_attn(s)[c] * x[b][c][n], s in 0..49, pad to 64
// ---------------------------------------------------------------------------
__global__ __launch_bounds__(256) void lspm_xc(const __bf16* __restrict__ xT,
                                               const float* __restrict__ wa1,
                                               const float* __restrict__ wa2,
                                               const float* __restrict__ wa3,
                                               const float* __restrict__ wa6,
                                               __bf16* __restrict__ xc) {
  size_t idx = (size_t)blockIdx.x * 256 + threadIdx.x;   // over B*64*N
  int n = idx % NT;
  int s = (idx / NT) % 64;
  int b = idx / ((size_t)NT * 64);
  if (s >= 50) { xc[idx] = (__bf16)0.f; return; }
  const float* wrow;
  if      (s == 0) wrow = wa1;
  else if (s < 5)  wrow = wa2 + (s - 1) * CCH;
  else if (s < 14) wrow = wa3 + (s - 5) * CCH;
  else             wrow = wa6 + (s - 14) * CCH;
  const __bf16* xr = xT + ((size_t)b * NT + n) * CCH;
  float acc = 0.f;
  for (int c = 0; c < CCH; ++c) acc += wrow[c] * (float)xr[c];
  xc[idx] = (__bf16)acc;
}

// ---------------------------------------------------------------------------
// 5. adaptive avg pool -> pooled[b][j][c], j = 50 cells over scales {1,2,3,6}
// ---------------------------------------------------------------------------
__global__ __launch_bounds__(256) void lspm_pool(const float* __restrict__ x,
                                                 float* __restrict__ pooled) {
  int idx = blockIdx.x * 256 + threadIdx.x;   // over B*50*C (c fastest)
  int c = idx % CCH;
  int j = (idx / CCH) % 50;
  int b = idx / (CCH * 50);
  int S, cell;
  if      (j == 0) { S = 1; cell = 0; }
  else if (j < 5)  { S = 2; cell = j - 1; }
  else if (j < 14) { S = 3; cell = j - 5; }
  else             { S = 6; cell = j - 14; }
  int win = 48 / S, sy = cell / S, sx = cell % S;
  const float* xb = x + ((size_t)b * CCH + c) * NT;
  float sum = 0.f;
  for (int hy = 0; hy < win; ++hy)
    for (int hx = 0; hx < win; ++hx)
      sum += xb[(sy * win + hy) * 48 + sx * win + hx];
  pooled[idx] = sum / (float)(win * win);
}

// 6. g_all[b][j][d] = relu(sum_c w_gap(scale(j))[d][c] * pooled[b][j][c])
__global__ __launch_bounds__(256) void lspm_g(const float* __restrict__ pooled,
                                              const float* __restrict__ wg1,
                                              const float* __restrict__ wg2,
                                              const float* __restrict__ wg3,
                                              const float* __restrict__ wg6,
                                              float* __restrict__ g_all) {
  int idx = blockIdx.x * 256 + threadIdx.x;   // over B*50*C (d fastest)
  int d = idx % CCH;
  int j = (idx / CCH) % 50;
  int b = idx / (CCH * 50);
  const float* wg;
  if      (j == 0) wg = wg1;
  else if (j < 5)  wg = wg2;
  else if (j < 14) wg = wg3;
  else             wg = wg6;
  const float* p = pooled + ((size_t)b * 50 + j) * CCH;
  const float* wr = wg + (size_t)d * CCH;
  float acc = 0.f;
  for (int c = 0; c < CCH; ++c) acc += wr[c] * p[c];
  g_all[idx] = fmaxf(acc, 0.f);
}

// ---------------------------------------------------------------------------
// 7. W_comb[b][d][0:256]  = sum_k w_final[d][k*C + j]           (identity paths)
//    W_comb[b][d][256+s]  = sum_c w_final[d][(k(s)+1)*C + c] * g_all[b][s][c]
//    (320-wide, padded with zeros)
// ---------------------------------------------------------------------------
__global__ __launch_bounds__(256) void lspm_wcomb(const float* __restrict__ wf,
                                                  const float* __restrict__ g_all,
                                                  __bf16* __restrict__ wcomb) {
  size_t idx = (size_t)blockIdx.x * 256 + threadIdx.x;   // over B*256*320
  int j = idx % KPAD;
  int d = (idx / KPAD) % CCH;
  int b = idx / ((size_t)KPAD * CCH);
  float v = 0.f;
  const float* wrow = wf + (size_t)d * (5 * CCH);
  if (j < CCH) {
    for (int k = 0; k < 5; ++k) v += wrow[k * CCH + j];
  } else if (j < 306) {
    int s = j - CCH;
    int k;
    if      (s == 0) k = 0;
    else if (s < 5)  k = 1;
    else if (s < 14) k = 2;
    else             k = 3;
    const float* wslice = wrow + (k + 1) * CCH;
    const float* g = g_all + ((size_t)b * 50 + s) * CCH;
    for (int c = 0; c < CCH; ++c) v += wslice[c] * g[c];
  }
  wcomb[idx] = (__bf16)v;
}

// ---------------------------------------------------------------------------
// 8. mm2: Z[b][256+s][m] = sum_n xc[b][s][n] * attn[b][n][m]
//    M=64 x N-tile=128, K=2304; attn tiles transposed via LDS for B-frags
// ---------------------------------------------------------------------------
__global__ __launch_bounds__(256) void lspm_mm2(const __bf16* __restrict__ xc,
                                                const __bf16* __restrict__ attn,
                                                __bf16* __restrict__ Z) {
  const int b  = blockIdx.z;
  const int m0 = blockIdx.x * 128;
  __shared__ __bf16 As[64][32];
  __shared__ __bf16 Bs[128][48];    // [col][k], padded stride
  const int t = threadIdx.x, wave = t >> 5, lane = t & 31;
  const int waveM = wave & 3, waveN = wave >> 2;
  const int half = lane >> 4, l = lane & 15;
  floatx8 acc[4];
  for (int i = 0; i < 4; ++i) acc[i] = (floatx8){0,0,0,0,0,0,0,0};
  const __bf16* xcb = xc + (size_t)b * 64 * NT;
  const __bf16* atb = attn + (size_t)b * NT * NT;

  for (int kk = 0; kk < NT; kk += 32) {
    { // A: 64x32
      int row = t >> 2, q = (t & 3) * 8;
      *(uint4*)(&As[row][q]) = *(const uint4*)(xcb + (size_t)row * NT + kk + q);
    }
    for (int i = 0; i < 2; ++i) { // B: 32 k-rows x 128 cols, transpose into Bs
      int s2 = t + i * 256;
      int k = s2 >> 4, cq = (s2 & 15) * 8;
      uint4 v = *(const uint4*)(atb + (size_t)(kk + k) * NT + m0 + cq);
      const __bf16* pv = (const __bf16*)&v;
      for (int e = 0; e < 8; ++e) Bs[cq + e][k] = pv[e];
    }
    __syncthreads();
    bf16x16 af = *(const bf16x16*)(&As[waveM * 16 + l][half * 16]);
    for (int ni = 0; ni < 4; ++ni) {
      bf16x16 bfr = *(const bf16x16*)(&Bs[waveN * 64 + ni * 16 + l][half * 16]);
      acc[ni] = WMMA_BF16(af, bfr, acc[ni]);
    }
    __syncthreads();
  }
  __bf16* Zb = Z + (size_t)b * KPAD * NT;
  const int rb = half * 8;
  for (int ni = 0; ni < 4; ++ni) {
    int col = m0 + waveN * 64 + ni * 16 + l;
    for (int r = 0; r < 8; ++r) {
      int row = waveM * 16 + rb + r;            // 0..63 -> Z rows 256..319
      Zb[(size_t)(CCH + row) * NT + col] = (__bf16)acc[ni][r];
    }
  }
}

// ---------------------------------------------------------------------------
// 9. final: out[b][d][n] = sum_j W_comb[b][d][j] * Z[b][j][n]   (256x320x2304)
// ---------------------------------------------------------------------------
__global__ __launch_bounds__(256) void lspm_final(const __bf16* __restrict__ wcomb,
                                                  const __bf16* __restrict__ Z,
                                                  float* __restrict__ out) {
  const int b  = blockIdx.z;
  const int d0 = blockIdx.y * 128;
  const int n0 = blockIdx.x * 128;
  __shared__ __bf16 As[128][32];
  __shared__ __bf16 Bs[128][48];
  const int t = threadIdx.x, wave = t >> 5, lane = t & 31;
  const int waveM = wave & 3, waveN = wave >> 2;
  const int half = lane >> 4, l = lane & 15;
  floatx8 acc[2][4];
  for (int i = 0; i < 2; ++i)
    for (int j = 0; j < 4; ++j) acc[i][j] = (floatx8){0,0,0,0,0,0,0,0};
  const __bf16* wcb = wcomb + (size_t)b * CCH * KPAD;
  const __bf16* Zb  = Z + (size_t)b * KPAD * NT;

  for (int kk = 0; kk < KPAD; kk += 32) {
    for (int i = 0; i < 2; ++i) { // A: 128x32 from W_comb (K contiguous)
      int slot = t + i * 256;
      int row = slot >> 2, q = (slot & 3) * 8;
      *(uint4*)(&As[row][q]) = *(const uint4*)(wcb + (size_t)(d0 + row) * KPAD + kk + q);
    }
    for (int i = 0; i < 2; ++i) { // B: 32 k-rows x 128 cols of Z, transpose
      int s2 = t + i * 256;
      int k = s2 >> 4, cq = (s2 & 15) * 8;
      uint4 v = *(const uint4*)(Zb + (size_t)(kk + k) * NT + n0 + cq);
      const __bf16* pv = (const __bf16*)&v;
      for (int e = 0; e < 8; ++e) Bs[cq + e][k] = pv[e];
    }
    __syncthreads();
    bf16x16 af[2], bfr[4];
    for (int mi = 0; mi < 2; ++mi)
      af[mi] = *(const bf16x16*)(&As[waveM * 32 + mi * 16 + l][half * 16]);
    for (int ni = 0; ni < 4; ++ni)
      bfr[ni] = *(const bf16x16*)(&Bs[waveN * 64 + ni * 16 + l][half * 16]);
    for (int mi = 0; mi < 2; ++mi)
      for (int ni = 0; ni < 4; ++ni)
        acc[mi][ni] = WMMA_BF16(af[mi], bfr[ni], acc[mi][ni]);
    __syncthreads();
  }
  float* ob = out + (size_t)b * CCH * NT;
  const int rb = half * 8;
  for (int mi = 0; mi < 2; ++mi)
    for (int ni = 0; ni < 4; ++ni) {
      int col = n0 + waveN * 64 + ni * 16 + l;
      for (int r = 0; r < 8; ++r) {
        int row = d0 + waveM * 32 + mi * 16 + rb + r;
        ob[(size_t)row * NT + col] = acc[mi][ni][r];
      }
    }
}

// ---------------------------------------------------------------------------
extern "C" void kernel_launch(void* const* d_in, const int* in_sizes, int n_in,
                              void* d_out, int out_size, void* d_ws, size_t ws_size,
                              hipStream_t stream) {
  const float* x   = (const float*)d_in[0];
  const float* wg1 = (const float*)d_in[1];
  const float* wa1 = (const float*)d_in[2];
  const float* wg2 = (const float*)d_in[3];
  const float* wa2 = (const float*)d_in[4];
  const float* wg3 = (const float*)d_in[5];
  const float* wa3 = (const float*)d_in[6];
  const float* wg6 = (const float*)d_in[7];
  const float* wa6 = (const float*)d_in[8];
  const float* wf  = (const float*)d_in[9];
  float* out = (float*)d_out;

  // workspace layout
  char* ws = (char*)d_ws;
  size_t off = 0;
  auto alloc = [&](size_t bytes) { char* p = ws + off; off += (bytes + 255) & ~(size_t)255; return p; };
  float*  scores = (float*)  alloc((size_t)NB * NT * NT * 4);   // 85 MB
  __bf16* attn   = (__bf16*) alloc((size_t)NB * NT * NT * 2);   // 42 MB
  __bf16* xT     = (__bf16*) alloc((size_t)NB * NT * CCH * 2);
  __bf16* Z      = (__bf16*) alloc((size_t)NB * KPAD * NT * 2);
  __bf16* xc     = (__bf16*) alloc((size_t)NB * 64 * NT * 2);
  float*  pooled = (float*)  alloc((size_t)NB * 50 * CCH * 4);
  float*  g_all  = (float*)  alloc((size_t)NB * 50 * CCH * 4);
  __bf16* wcomb  = (__bf16*) alloc((size_t)NB * CCH * KPAD * 2);
  (void)ws_size; (void)in_sizes; (void)n_in; (void)out_size;

  lspm_convert<<<(NB * CCH * NT) / 256, 256, 0, stream>>>(x, xT, Z);
  lspm_gram   <<<dim3(NT / 128, NT / 128, NB), 256, 0, stream>>>(xT, scores);
  lspm_softmax<<<NB * NT, 256, 0, stream>>>(scores, attn);
  lspm_xc     <<<(NB * 64 * NT) / 256, 256, 0, stream>>>(xT, wa1, wa2, wa3, wa6, xc);
  lspm_pool   <<<(NB * 50 * CCH) / 256, 256, 0, stream>>>(x, pooled);
  lspm_g      <<<(NB * 50 * CCH) / 256, 256, 0, stream>>>(pooled, wg1, wg2, wg3, wg6, g_all);
  lspm_wcomb  <<<(NB * CCH * KPAD) / 256, 256, 0, stream>>>(wf, g_all, wcomb);
  lspm_mm2    <<<dim3(NT / 128, 1, NB), 256, 0, stream>>>(xc, attn, Z);
  lspm_final  <<<dim3(NT / 128, CCH / 128, NB), 256, 0, stream>>>(wcomb, Z, out);
}
